// GINE_LSPE_19069654794758
// MI455X (gfx1250) — matrix operand order repaired and checked by reading, not compiled, not run on hardware
//
#include <hip/hip_runtime.h>
#include <hip/hip_bf16.h>

// ---------------- problem constants (match reference) ----------------
#define NN    50000      // nodes
#define NNP   50048      // nodes padded to 64-row GEMM blocks (782*64)
#define EE    1600000    // edges
#define DD    128        // hidden dim
#define PP    20         // LSPE dim
#define LL    4          // layers
#define GG    512        // graphs
#define DP_   148        // D + P
#define DPAD  160        // padded D+P: multiple of 32 (K) and 16 (N)
#define KT    5          // K tiles of 32 over DPAD
#define NROWT 4          // row tiles per block
#define BROWS 64         // rows per block (NROWT*16)
#define CT1   10         // col tiles, GEMM1 (160 cols)
#define CT2   8          // col tiles, GEMM2 (128 cols)

typedef __attribute__((ext_vector_type(16))) __bf16 v16bf;
typedef __attribute__((ext_vector_type(8)))  float  v8f;
typedef __attribute__((ext_vector_type(4)))  int    v4i_t;

union FRAG { uint4 q[2]; v16bf v; };

// ---- optional gfx1250 async global->LDS staging (guarded; falls back to uint4 copy) ----
#if defined(__has_builtin)
#if __has_builtin(__builtin_amdgcn_global_load_async_to_lds_b128)
#define HAVE_ASYNC_LDS 1
#endif
#endif

static __device__ __forceinline__ void wait_async0() {
#if defined(__has_builtin)
#if __has_builtin(__builtin_amdgcn_s_wait_asynccnt)
    __builtin_amdgcn_s_wait_asynccnt(0);
    return;
#else
    asm volatile("s_wait_asynccnt 0" ::: "memory");
    return;
#endif
#else
    asm volatile("s_wait_asynccnt 0" ::: "memory");
#endif
}

static __device__ __forceinline__ unsigned short f2bf(float f) {
    unsigned int u = __float_as_uint(f);
    unsigned int r = (u + 0x7FFFu + ((u >> 16) & 1u)) >> 16;   // RNE
    return (unsigned short)r;
}

// ---------------- init: hp = [node_emb[x] || p || 0pad], zero pad rows ----------------
__global__ void init_hp_kernel(const int* __restrict__ x, const float* __restrict__ p,
                               const float* __restrict__ node_emb, float* __restrict__ hp) {
    long long idx = (long long)blockIdx.x * blockDim.x + threadIdx.x;
    if (idx >= (long long)NNP * DPAD) return;
    int i = (int)(idx / DPAD), f = (int)(idx % DPAD);
    float v = 0.f;
    if (i < NN) {
        if (f < DD)       v = node_emb[x[i] * DD + f];
        else if (f < DP_) v = p[(long long)i * PP + (f - DD)];
    }
    hp[idx] = v;
}

// ---------------- weight prep: pack B into WMMA-fragment order ----------------
// Bpack layout: [l][colTile][kTile][lane(32)][elem(16)]
// lane: n = ct*16 + (lane&15), kgroup = lane>>4 ; elem j: k = kt*32 + kgroup*16 + j
__global__ void pack_w1_kernel(const float* __restrict__ w, unsigned short* __restrict__ o) {
    int idx = blockIdx.x * blockDim.x + threadIdx.x;
    if (idx >= LL * CT1 * KT * 32 * 16) return;
    int j    = idx & 15;
    int lane = (idx >> 4) & 31;
    int kt   = (idx / (16 * 32)) % KT;
    int ct   = (idx / (16 * 32 * KT)) % CT1;
    int l    =  idx / (16 * 32 * KT * CT1);
    int k = kt * 32 + (lane >> 4) * 16 + j;
    int n = ct * 16 + (lane & 15);
    float v = (k < DP_ && n < DP_) ? w[((long long)l * DP_ + k) * DP_ + n] : 0.f;
    o[idx] = f2bf(v);
}
__global__ void pack_w2_kernel(const float* __restrict__ w, unsigned short* __restrict__ o) {
    int idx = blockIdx.x * blockDim.x + threadIdx.x;
    if (idx >= LL * CT2 * KT * 32 * 16) return;
    int j    = idx & 15;
    int lane = (idx >> 4) & 31;
    int kt   = (idx / (16 * 32)) % KT;
    int ct   = (idx / (16 * 32 * KT)) % CT2;
    int l    =  idx / (16 * 32 * KT * CT2);
    int k = kt * 32 + (lane >> 4) * 16 + j;
    int n = ct * 16 + (lane & 15);                 // always < 128
    float v = (k < DP_) ? w[((long long)l * DP_ + k) * DD + n] : 0.f;
    o[idx] = f2bf(v);
}
__global__ void pad_vec_kernel(const float* __restrict__ in, float* __restrict__ out) {
    int idx = blockIdx.x * blockDim.x + threadIdx.x;
    if (idx >= LL * DPAD) return;
    int l = idx / DPAD, f = idx % DPAD;
    out[idx] = (f < DP_) ? in[l * DP_ + f] : 0.f;
}
// eproj[l][a][f] = (edge_emb[a] @ le_W[l])[f] + le_b[l][f]   (4 rows per layer)
__global__ void eproj_kernel(const float* __restrict__ edge_emb, const float* __restrict__ leW,
                             const float* __restrict__ leb, float* __restrict__ eproj) {
    int idx = blockIdx.x * blockDim.x + threadIdx.x;
    if (idx >= LL * 4 * DPAD) return;
    int l = idx / (4 * DPAD), a = (idx / DPAD) % 4, f = idx % DPAD;
    float s = 0.f;
    if (f < DP_) {
        s = leb[l * DP_ + f];
        const float* W = leW + (long long)l * DD * DP_;
        const float* e = edge_emb + a * DD;
        for (int k = 0; k < DD; ++k) s += e[k] * W[k * DP_ + f];
    }
    eproj[idx] = s;
}

// ---------------- edge message + scatter (wave per edge) ----------------
__global__ void edge_message_kernel(const int* __restrict__ ei, const int* __restrict__ ea,
                                    const float* __restrict__ hp, const float* __restrict__ eproj,
                                    float* __restrict__ agg, float* __restrict__ aggp) {
    int wid  = blockIdx.x * (blockDim.x >> 5) + (threadIdx.x >> 5);
    int lane = threadIdx.x & 31;
    if (wid >= EE) return;
    int src = ei[wid], dst = ei[EE + wid], attr = ea[wid];
    const float* hrow = hp + (long long)src * DPAD;
    const float* ep   = eproj + attr * DPAD;
    float*       arow = agg + (long long)dst * DPAD;
    for (int f = lane; f < DP_; f += 32) {
        float m = hrow[f] + ep[f];
        m = m > 0.f ? m : 0.f;
        atomicAdd(arow + f, m);
    }
    if (lane < PP) atomicAdd(aggp + (long long)dst * 32 + lane, hrow[DD + lane]);
}

// ---------------- (hp+agg) -> bf16 A operand ----------------
__global__ void fuse_add_convert_kernel(const float* __restrict__ hp, const float* __restrict__ agg,
                                        unsigned short* __restrict__ A) {
    long long idx = (long long)blockIdx.x * blockDim.x + threadIdx.x;
    if (idx >= (long long)NNP * DPAD) return;
    A[idx] = f2bf(hp[idx] + agg[idx]);
}

// ---------------- WMMA GEMM: C[64-row block x ncols] = A @ Bpack + bias
// 64 A rows staged in LDS (async global->LDS path when available); each wave
// owns one 16-col tile and 4 accumulators (20 WMMAs per wave).
__global__ __launch_bounds__(320) void wmma_gemm_kernel(
        const unsigned short* __restrict__ A,     // [NNP x DPAD] bf16
        const unsigned short* __restrict__ Bp,    // packed [ct][kt][32][16] bf16
        const float* __restrict__ bias, float* __restrict__ C,
        int ldc, int residual) {
    __shared__ __align__(16) unsigned short As[BROWS * DPAD];
    const int row0 = blockIdx.x * BROWS;

    // stage A tile: 64x160 bf16 = 1280 x 16B
#if defined(HAVE_ASYNC_LDS)
    {
        const unsigned short* gsrc = A + (long long)row0 * DPAD;
        for (int t = threadIdx.x; t < BROWS * DPAD / 8; t += blockDim.x) {
            __builtin_amdgcn_global_load_async_to_lds_b128(
                (__attribute__((address_space(1))) v4i_t*)(gsrc + (long long)t * 8),
                (__attribute__((address_space(3))) v4i_t*)(As + (long long)t * 8),
                0, 0);
        }
        wait_async0();
    }
#else
    {
        const uint4* gsrc = (const uint4*)(A + (long long)row0 * DPAD);
        uint4* ldst = (uint4*)As;
        for (int t = threadIdx.x; t < BROWS * DPAD / 8; t += blockDim.x)
            ldst[t] = gsrc[t];
    }
#endif
    __syncthreads();

    const int lane = threadIdx.x & 31;
    const int ct   = threadIdx.x >> 5;
    const int col0 = ct * 16;
    const int m16  = lane & 15;
    const int hi   = lane >> 4;

    const float bv = bias[col0 + m16];
    v8f acc[NROWT];
#pragma unroll
    for (int rt = 0; rt < NROWT; ++rt)
#pragma unroll
        for (int i = 0; i < 8; ++i) acc[rt][i] = bv;

#pragma unroll
    for (int kt = 0; kt < KT; ++kt) {
        const int k0 = kt * 32;
        // B fragment: two 16B loads from packed layout
        FRAG fb;
        const unsigned short* bp = Bp + (((long long)ct * KT + kt) * 32 + lane) * 16;
        fb.q[0] = *(const uint4*)(bp);
        fb.q[1] = *(const uint4*)(bp + 8);
#pragma unroll
        for (int rt = 0; rt < NROWT; ++rt) {
            // A fragment: elems 0..15 = k0+hi*16.. (contiguous)
            const unsigned short* ap = &As[(rt * 16 + m16) * DPAD + k0 + hi * 16];
            FRAG fa;
            fa.q[0] = *(const uint4*)(ap);
            fa.q[1] = *(const uint4*)(ap + 8);
            acc[rt] = __builtin_amdgcn_wmma_f32_16x16x32_bf16(false, fa.v, false, fb.v,
                                                              (short)0, acc[rt], false, false);
        }
    }
    // C layout: VGPR i -> M = i + 8*hi, N = lane&15
#pragma unroll
    for (int rt = 0; rt < NROWT; ++rt)
#pragma unroll
        for (int i = 0; i < 8; ++i) {
            long long off = (long long)(row0 + rt * 16 + i + hi * 8) * ldc + (col0 + m16);
            if (residual) C[off] += acc[rt][i];
            else          C[off]  = acc[rt][i];
        }
}

// ---------------- batchnorm column stats (over the NN real rows) ----------------
__global__ void bn_stats_kernel(const float* __restrict__ z, float* __restrict__ mean,
                                float* __restrict__ rstd) {
    const int col = blockIdx.x;
    float s = 0.f, s2 = 0.f;
    for (int r = threadIdx.x; r < NN; r += blockDim.x) {
        float v = z[(long long)r * DPAD + col];
        s += v; s2 += v * v;
    }
    __shared__ float sh[256], sh2[256];
    sh[threadIdx.x] = s; sh2[threadIdx.x] = s2;
    __syncthreads();
    for (int st = 128; st > 0; st >>= 1) {
        if (threadIdx.x < st) { sh[threadIdx.x] += sh[threadIdx.x + st];
                                sh2[threadIdx.x] += sh2[threadIdx.x + st]; }
        __syncthreads();
    }
    if (threadIdx.x == 0) {
        float mu  = sh[0] / (float)NN;
        float var = sh2[0] / (float)NN - mu * mu;
        mean[col] = mu;
        rstd[col] = rsqrtf(var + 1e-5f);
    }
}

__global__ void bn_relu_convert_kernel(const float* __restrict__ z, const float* __restrict__ gamma,
                                       const float* __restrict__ beta, const float* __restrict__ mean,
                                       const float* __restrict__ rstd, unsigned short* __restrict__ A2) {
    long long idx = (long long)blockIdx.x * blockDim.x + threadIdx.x;
    if (idx >= (long long)NNP * DPAD) return;
    int col = (int)(idx % DPAD);
    float v = gamma[col] * (z[idx] - mean[col]) * rstd[col] + beta[col];
    v = v > 0.f ? v : 0.f;
    A2[idx] = f2bf(v);
}

// ---------------- LSPE p update (wave per node) ----------------
__global__ void p_update_kernel(const float* __restrict__ aggp, const float* __restrict__ pW1,
                                const float* __restrict__ pb1, const float* __restrict__ pW2,
                                const float* __restrict__ pb2, float* __restrict__ hp) {
    __shared__ float pn[8][PP];
    __shared__ float t1[8][PP];
    const int wv = threadIdx.x >> 5, lane = threadIdx.x & 31;
    const int i = blockIdx.x * 8 + wv;          // NN = 6250 * 8, exact
    if (lane < PP) pn[wv][lane] = hp[(long long)i * DPAD + DD + lane] + aggp[(long long)i * 32 + lane];
    __syncthreads();
    if (lane < PP) {
        float s = pb1[lane];
        for (int k = 0; k < PP; ++k) s += pn[wv][k] * pW1[k * PP + lane];
        t1[wv][lane] = s > 0.f ? s : 0.f;
    }
    __syncthreads();
    if (lane < PP) {
        float s = pb2[lane];
        for (int k = 0; k < PP; ++k) s += t1[wv][k] * pW2[k * PP + lane];
        hp[(long long)i * DPAD + DD + lane] += tanhf(s);
    }
}

// ---------------- global mean pool ----------------
__global__ void pool_kernel(const float* __restrict__ hp, const int* __restrict__ bidx,
                            float* __restrict__ sums, float* __restrict__ cnt) {
    long long idx = (long long)blockIdx.x * blockDim.x + threadIdx.x;
    if (idx >= (long long)NN * DP_) return;
    int i = (int)(idx / DP_), f = (int)(idx % DP_);
    int b = bidx[i];
    atomicAdd(&sums[(long long)b * DPAD + f], hp[(long long)i * DPAD + f]);
    if (f == 0) atomicAdd(&cnt[b], 1.f);
}

// ---------------- readout MLP (wave per graph) ----------------
__global__ void mlp_kernel(const float* __restrict__ sums, const float* __restrict__ cnt,
                           const float* __restrict__ W1, const float* __restrict__ b1,
                           const float* __restrict__ W2, const float* __restrict__ b2,
                           const float* __restrict__ W3, const float* __restrict__ b3,
                           float* __restrict__ out) {
    __shared__ float gsh[8][DP_];
    __shared__ float z1[8][74];
    __shared__ float z2[8][37];
    const int wv = threadIdx.x >> 5, lane = threadIdx.x & 31;
    const int g = blockIdx.x * 8 + wv;          // GG = 64 * 8, exact
    float c = cnt[g]; c = c > 1.f ? c : 1.f;
    for (int f = lane; f < DP_; f += 32) gsh[wv][f] = sums[(long long)g * DPAD + f] / c;
    __syncthreads();
    for (int j = lane; j < 74; j += 32) {
        float s = b1[j];
        for (int k = 0; k < DP_; ++k) s += gsh[wv][k] * W1[k * 74 + j];
        z1[wv][j] = s > 0.f ? s : 0.f;
    }
    __syncthreads();
    for (int j = lane; j < 37; j += 32) {
        float s = b2[j];
        for (int k = 0; k < 74; ++k) s += z1[wv][k] * W2[k * 37 + j];
        z2[wv][j] = s > 0.f ? s : 0.f;
    }
    __syncthreads();
    if (lane == 0) {
        float s = b3[0];
        for (int k = 0; k < 37; ++k) s += z2[wv][k] * W3[k];
        out[g] = s;
    }
}

// =====================================================================
extern "C" void kernel_launch(void* const* d_in, const int* in_sizes, int n_in,
                              void* d_out, int out_size, void* d_ws, size_t ws_size,
                              hipStream_t stream) {
    const int*   x        = (const int*)  d_in[0];
    const int*   ei       = (const int*)  d_in[1];
    const int*   ea       = (const int*)  d_in[2];
    const int*   bidx     = (const int*)  d_in[3];
    const float* p_in     = (const float*)d_in[4];
    const float* node_emb = (const float*)d_in[5];
    const float* edge_emb = (const float*)d_in[6];
    const float* le_W     = (const float*)d_in[7];
    const float* le_b     = (const float*)d_in[8];
    const float* h_W1     = (const float*)d_in[9];
    const float* h_b1     = (const float*)d_in[10];
    const float* h_gamma  = (const float*)d_in[11];
    const float* h_beta   = (const float*)d_in[12];
    const float* h_W2     = (const float*)d_in[13];
    const float* h_b2     = (const float*)d_in[14];
    const float* p_W1     = (const float*)d_in[15];
    const float* p_b1     = (const float*)d_in[16];
    const float* p_W2     = (const float*)d_in[17];
    const float* p_b2     = (const float*)d_in[18];
    const float* m_W1     = (const float*)d_in[19];
    const float* m_b1     = (const float*)d_in[20];
    const float* m_W2     = (const float*)d_in[21];
    const float* m_b2     = (const float*)d_in[22];
    const float* m_W3     = (const float*)d_in[23];
    const float* m_b3     = (const float*)d_in[24];
    float* out = (float*)d_out;

    // ---- carve workspace ----
    char* base = (char*)d_ws; size_t off = 0;
    auto carve = [&](size_t bytes) { void* q = base + off; off += (bytes + 255) & ~size_t(255); return q; };
    float*          hp    = (float*)         carve((size_t)NNP * DPAD * 4);
    float*          agg   = (float*)         carve((size_t)NNP * DPAD * 4);
    float*          zbuf  = (float*)         carve((size_t)NNP * DPAD * 4);
    float*          aggp  = (float*)         carve((size_t)NN * 32 * 4);
    unsigned short* Abf   = (unsigned short*)carve((size_t)NNP * DPAD * 2);
    unsigned short* A2bf  = (unsigned short*)carve((size_t)NNP * DPAD * 2);
    unsigned short* Bp1   = (unsigned short*)carve((size_t)LL * CT1 * KT * 32 * 16 * 2);
    unsigned short* Bp2   = (unsigned short*)carve((size_t)LL * CT2 * KT * 32 * 16 * 2);
    float*          b1p   = (float*)         carve((size_t)LL * DPAD * 4);
    float*          gammp = (float*)         carve((size_t)LL * DPAD * 4);
    float*          betap = (float*)         carve((size_t)LL * DPAD * 4);
    float*          eproj = (float*)         carve((size_t)LL * 4 * DPAD * 4);
    float*          meanb = (float*)         carve((size_t)DPAD * 4);
    float*          rstdb = (float*)         carve((size_t)DPAD * 4);
    float*          sums  = (float*)         carve((size_t)GG * DPAD * 4);
    float*          cntb  = (float*)         carve((size_t)GG * 4);
    (void)ws_size; (void)in_sizes; (void)n_in; (void)out_size;

    const int TB = 256;
    const long long elemsHP = (long long)NNP * DPAD;
    const int gHP = (int)((elemsHP + TB - 1) / TB);
    const int nPack1 = LL * CT1 * KT * 32 * 16;
    const int nPack2 = LL * CT2 * KT * 32 * 16;

    // ---- prep ----
    init_hp_kernel<<<gHP, TB, 0, stream>>>(x, p_in, node_emb, hp);
    pack_w1_kernel<<<(nPack1 + TB - 1) / TB, TB, 0, stream>>>(h_W1, Bp1);
    pack_w2_kernel<<<(nPack2 + TB - 1) / TB, TB, 0, stream>>>(h_W2, Bp2);
    pad_vec_kernel<<<(LL * DPAD + TB - 1) / TB, TB, 0, stream>>>(h_b1, b1p);
    pad_vec_kernel<<<(LL * DPAD + TB - 1) / TB, TB, 0, stream>>>(h_gamma, gammp);
    pad_vec_kernel<<<(LL * DPAD + TB - 1) / TB, TB, 0, stream>>>(h_beta, betap);
    eproj_kernel<<<(LL * 4 * DPAD + TB - 1) / TB, TB, 0, stream>>>(edge_emb, le_W, le_b, eproj);

    // ---- layers ----
    for (int l = 0; l < LL; ++l) {
        (void)hipMemsetAsync(agg,  0, (size_t)NNP * DPAD * 4, stream);
        (void)hipMemsetAsync(aggp, 0, (size_t)NN * 32 * 4, stream);
        edge_message_kernel<<<EE / 8, TB, 0, stream>>>(ei, ea, hp, eproj + (size_t)l * 4 * DPAD,
                                                       agg, aggp);
        fuse_add_convert_kernel<<<gHP, TB, 0, stream>>>(hp, agg, Abf);
        // z = (hp+agg) @ W1 + b1      [50048 x 160] x [160 x 160]
        wmma_gemm_kernel<<<NNP / BROWS, CT1 * 32, 0, stream>>>(
            Abf, Bp1 + (size_t)l * CT1 * KT * 32 * 16,
            b1p + (size_t)l * DPAD, zbuf, DPAD, 0);
        bn_stats_kernel<<<DPAD, 256, 0, stream>>>(zbuf, meanb, rstdb);
        bn_relu_convert_kernel<<<gHP, TB, 0, stream>>>(zbuf, gammp + (size_t)l * DPAD,
                                                       betap + (size_t)l * DPAD, meanb, rstdb, A2bf);
        // hp[:, :128] += relu(z_bn) @ W2 + b2   [50048 x 160] x [160 x 128]
        wmma_gemm_kernel<<<NNP / BROWS, CT2 * 32, 0, stream>>>(
            A2bf, Bp2 + (size_t)l * CT2 * KT * 32 * 16,
            h_b2 + (size_t)l * DD, hp, DPAD, 1);
        p_update_kernel<<<NN / 8, 256, 0, stream>>>(aggp, p_W1 + (size_t)l * PP * PP,
                                                    p_b1 + (size_t)l * PP,
                                                    p_W2 + (size_t)l * PP * PP,
                                                    p_b2 + (size_t)l * PP, hp);
    }

    // ---- pool + readout ----
    (void)hipMemsetAsync(sums, 0, (size_t)GG * DPAD * 4, stream);
    (void)hipMemsetAsync(cntb, 0, (size_t)GG * 4, stream);
    const long long elemsPool = (long long)NN * DP_;
    pool_kernel<<<(int)((elemsPool + TB - 1) / TB), TB, 0, stream>>>(hp, bidx, sums, cntb);
    mlp_kernel<<<GG / 8, 256, 0, stream>>>(sums, cntb, m_W1, m_b1, m_W2, m_b2, m_W3, m_b3, out);
}
